// ConvLayer_82497731822219
// MI455X (gfx1250) — matrix operand order repaired and checked by reading, not compiled, not run on hardware
//
#include <hip/hip_runtime.h>

typedef __attribute__((ext_vector_type(16))) __bf16 v16bf;
typedef __attribute__((ext_vector_type(8)))  float  v8f;

#define B_   4
#define H_   128
#define W_   128
#define HW_  (128*128)

__device__ __forceinline__ __bf16 f2bf(float f) {
  union { float f; unsigned u; } in; in.f = f;
  unsigned u = in.u;
  unsigned r = (u + 0x7FFFu + ((u >> 16) & 1u)) >> 16;  // round-to-nearest-even
  union { unsigned short s; __bf16 b; } out; out.s = (unsigned short)r;
  return out.b;
}

// ---------------------------------------------------------------- fp32 -> bf16
__global__ void wcvt_kernel(const float* __restrict__ w, __bf16* __restrict__ wbf, int n) {
  int i = blockIdx.x * 256 + threadIdx.x;
  if (i < n) wbf[i] = f2bf(w[i]);
}

// ------------------------------------------------- dense 3x3 conv, C -> 27 ch
template<int C>
__global__ void __launch_bounds__(128)
offconv_kernel(const float* __restrict__ x, const float* __restrict__ wo,
               const float* __restrict__ bo, float* __restrict__ om) {
  __shared__ float s_w[16 * 243];                // 16 channels x 27 out x 9 taps
  const int b = blockIdx.x / H_;
  const int h = blockIdx.x % H_;
  const int w = threadIdx.x;                     // 128 threads = one row
  float acc[27];
#pragma unroll
  for (int o = 0; o < 27; o++) acc[o] = 0.f;

  for (int c0 = 0; c0 < C; c0 += 16) {
    __syncthreads();
    for (int e = threadIdx.x; e < 16 * 243; e += 128) {
      int cl = e / 243, rem = e % 243;
      int o = rem / 9, j = rem % 9;
      s_w[e] = wo[(o * C + (c0 + cl)) * 9 + j];
    }
    __syncthreads();
    for (int cl = 0; cl < 16; cl++) {
      const float* xp = x + (size_t)(b * C + c0 + cl) * HW_;
      float xv[9];
#pragma unroll
      for (int j = 0; j < 9; j++) {
        int yy = h + j / 3 - 1, xx = w + j % 3 - 1;
        xv[j] = (yy >= 0 && yy < H_ && xx >= 0 && xx < W_) ? xp[yy * W_ + xx] : 0.f;
      }
      const float* sw = &s_w[cl * 243];
#pragma unroll
      for (int o = 0; o < 27; o++)
#pragma unroll
        for (int j = 0; j < 9; j++) acc[o] += xv[j] * sw[o * 9 + j];
    }
  }
#pragma unroll
  for (int o = 0; o < 27; o++)
    om[(size_t)(b * 27 + o) * HW_ + h * W_ + w] = acc[o] + bo[o];
}

// -------------------- fused deformable sampling + bf16 WMMA GEMM (+bias)
// grid.x = B*H (one row of 128 pixels), grid.y = OTOT/128, block = 256 (8 waves)
// Double-buffered LDS tiles: one barrier per K-step; all 8 A-fragments are
// preloaded to registers so the 8 WMMAs issue back-to-back after one dscnt wait.
template<int C, int OTOT>
__global__ void __launch_bounds__(256)
dcn_kernel(const float* __restrict__ x, const float* __restrict__ om,
           const __bf16* __restrict__ wbf, const float* __restrict__ bias,
           float* __restrict__ out) {
  constexpr int KK = C * 9;
  __shared__ int    s_idx[9][128][4];   // clamped spatial offsets
  __shared__ float  s_wt [9][128][4];   // bilinear weight * validity * mask
  __shared__ __bf16 sA[2][128][32];     // weights tile  (O x K), row = 64B
  __shared__ __bf16 sB[2][128][32];     // sampled tile  (N x K), row = 64B

  const int b      = blockIdx.x / H_;
  const int h      = blockIdx.x % H_;
  const int o_base = blockIdx.y * 128;
  const int tid    = threadIdx.x;
  const int lane   = tid & 31;
  const int wave   = tid >> 5;
  const int khalf  = lane >> 4;

  // ---- Phase 1: per-(tap,pixel) sampling tables (shared across all channels)
  for (int e = tid; e < 9 * 128; e += 256) {
    int k = e >> 7, n = e & 127;
    float offy = om[(size_t)(b * 27 +      k) * HW_ + h * W_ + n];
    float offx = om[(size_t)(b * 27 +  9 + k) * HW_ + h * W_ + n];
    float ml   = om[(size_t)(b * 27 + 18 + k) * HW_ + h * W_ + n];
    float mask = 1.f / (1.f + __expf(-ml));
    float py = (float)h + (float)(k / 3 - 1) + offy;
    float px = (float)n + (float)(k % 3 - 1) + offx;
    float y0f = floorf(py), x0f = floorf(px);
    float ly = py - y0f, lx = px - x0f;
    int y0 = (int)y0f, x0 = (int)x0f;
    float wq[4] = { (1.f - ly) * (1.f - lx), (1.f - ly) * lx,
                    ly * (1.f - lx),          ly * lx };
    const int dy[4] = {0, 0, 1, 1}, dx[4] = {0, 1, 0, 1};
#pragma unroll
    for (int q = 0; q < 4; q++) {
      int yi = y0 + dy[q], xi = x0 + dx[q];
      bool valid = (yi >= 0) && (yi < H_) && (xi >= 0) && (xi < W_);
      int yc = min(max(yi, 0), H_ - 1), xc = min(max(xi, 0), W_ - 1);
      s_idx[k][n][q] = yc * W_ + xc;
      s_wt [k][n][q] = valid ? wq[q] * mask : 0.f;
    }
  }

  // stage one K tile (32 (c,k) pairs) into LDS buffer `buf`
  auto stage = [&](int buf, int kbase) {
    {   // weights: 128 rows x 32 bf16, 16 bf16 per thread, b128 loads
      int o = tid >> 1, half = tid & 1;
      const uint4* src = (const uint4*)(wbf + (size_t)(o_base + o) * KK + kbase + half * 16);
      uint4* dst = (uint4*)&sA[buf][o][half * 16];
      dst[0] = src[0];
      dst[1] = src[1];
    }
    // sampled activations: 32 K x 128 pixels, 4 gathers + lerp each
    for (int e = tid; e < 32 * 128; e += 256) {
      int kk = e >> 7, n = e & 127;
      int K = kbase + kk;
      int c = K / 9, k = K % 9;
      const float* xp = x + (size_t)(b * C + c) * HW_;
      const int*   ip = s_idx[k][n];
      const float* wp = s_wt [k][n];
      float v = wp[0] * xp[ip[0]] + wp[1] * xp[ip[1]] +
                wp[2] * xp[ip[2]] + wp[3] * xp[ip[3]];
      sB[buf][n][kk] = f2bf(v);
    }
  };

  __syncthreads();          // sampling tables ready
  stage(0, 0);              // prologue: fill buffer 0

  v8f zero = {};
  v8f acc[8];
#pragma unroll
  for (int t = 0; t < 8; t++) acc[t] = zero;

  union Frag { uint4 q[2]; v16bf v; };

  // ---- Phase 2: K loop, one barrier per step, double-buffered
  int p = 0;
  for (int kk0 = 0; kk0 < KK; kk0 += 32, p ^= 1) {
    __syncthreads();        // buffer p staged by everyone

    // B fragment: lane = column n, lanes 0-15 -> K 0..15, lanes 16-31 -> K 16..31
    Frag fb;
    {
      int n = wave * 16 + (lane & 15);
      const uint4* q = (const uint4*)&sB[p][n][khalf * 16];
      fb.q[0] = q[0]; fb.q[1] = q[1];
    }
    // A fragments for all 8 O-tiles: lane = row M,
    // K range kh*16..kh*16+15
    Frag fa[8];
#pragma unroll
    for (int t = 0; t < 8; t++) {
      int o = t * 16 + (lane & 15);
      fa[t].q[0] = *(const uint4*)&sA[p][o][khalf * 16];
      fa[t].q[1] = *(const uint4*)&sA[p][o][khalf * 16 + 8];
    }
    // 8 back-to-back WMMAs (independent accumulators)
#pragma unroll
    for (int t = 0; t < 8; t++)
      acc[t] = __builtin_amdgcn_wmma_f32_16x16x32_bf16(
          false, fa[t].v, false, fb.v, (short)0, acc[t], false, false);

    // stage next K tile into the other buffer while WMMAs drain
    int kn = kk0 + 32;
    if (kn < KK) stage(p ^ 1, kn);
  }

  // ---- writeback: C-matrix layout — lane = N (lane&15), VGPR r = M (r + 8*khalf)
  const int npix = wave * 16 + (lane & 15);
#pragma unroll
  for (int t = 0; t < 8; t++)
#pragma unroll
    for (int r = 0; r < 8; r++) {
      int og = o_base + t * 16 + r + 8 * khalf;
      out[(size_t)(b * OTOT + og) * HW_ + h * W_ + npix] = acc[t][r] + bias[og];
    }
}

// ---------------------------------------------------------------- BN stats
template<int O>
__global__ void __launch_bounds__(256)
bnstats_kernel(const float* __restrict__ y, float* __restrict__ stats) {
  __shared__ float s_s[256], s_q[256];
  int o = blockIdx.x;
  float s = 0.f, q = 0.f;
  for (int b = 0; b < B_; b++) {
    const float* p = y + (size_t)(b * O + o) * HW_;
    for (int i = threadIdx.x; i < HW_; i += 256) { float v = p[i]; s += v; q += v * v; }
  }
  s_s[threadIdx.x] = s; s_q[threadIdx.x] = q;
  __syncthreads();
  for (int st = 128; st > 0; st >>= 1) {
    if (threadIdx.x < st) {
      s_s[threadIdx.x] += s_s[threadIdx.x + st];
      s_q[threadIdx.x] += s_q[threadIdx.x + st];
    }
    __syncthreads();
  }
  if (threadIdx.x == 0) { stats[2 * o] = s_s[0]; stats[2 * o + 1] = s_q[0]; }
}

// ------------------------------------------------------- BN apply + ReLU (in place)
template<int O>
__global__ void __launch_bounds__(256)
bnapply_kernel(float* __restrict__ y, const float* __restrict__ stats,
               const float* __restrict__ g, const float* __restrict__ be) {
  const float inv_n = 1.f / (float)(B_ * HW_);
  size_t i = (size_t)blockIdx.x * 256 + threadIdx.x;
  if (i >= (size_t)B_ * O * HW_) return;
  int o = (int)((i / HW_) % O);
  float mu  = stats[2 * o] * inv_n;
  float var = stats[2 * o + 1] * inv_n - mu * mu;
  float v = (y[i] - mu) * rsqrtf(var + 1e-5f);
  v = v * g[o] + be[o];
  y[i] = v > 0.f ? v : 0.f;
}

extern "C" void kernel_launch(void* const* d_in, const int* in_sizes, int n_in,
                              void* d_out, int out_size, void* d_ws, size_t ws_size,
                              hipStream_t stream) {
  const float* x      = (const float*)d_in[0];
  const float* w_off1 = (const float*)d_in[1];
  const float* b_off1 = (const float*)d_in[2];
  const float* w1     = (const float*)d_in[3];
  const float* b1     = (const float*)d_in[4];
  const float* g1     = (const float*)d_in[5];
  const float* be1    = (const float*)d_in[6];
  const float* w_off2 = (const float*)d_in[7];
  const float* b_off2 = (const float*)d_in[8];
  const float* w2     = (const float*)d_in[9];
  const float* b2     = (const float*)d_in[10];
  const float* g2     = (const float*)d_in[11];
  const float* be2    = (const float*)d_in[12];
  float* out = (float*)d_out;
  (void)in_sizes; (void)n_in; (void)out_size; (void)ws_size;

  char* ws = (char*)d_ws;
  size_t off = 0;
  auto alloc = [&](size_t bytes) -> char* {
    char* p = ws + off; off += (bytes + 255) & ~(size_t)255; return p;
  };
  float*  om    = (float*) alloc((size_t)B_ * 27  * HW_ * 4);   // offset-conv output (reused)
  float*  y1    = (float*) alloc((size_t)B_ * 128 * HW_ * 4);   // layer-1 activations
  __bf16* wbf1  = (__bf16*)alloc((size_t)128 * 2304 * 2);
  __bf16* wbf2  = (__bf16*)alloc((size_t)256 * 1152 * 2);
  float*  stats = (float*) alloc(2 * 256 * 4);

  wcvt_kernel<<<(128 * 2304 + 255) / 256, 256, 0, stream>>>(w1, wbf1, 128 * 2304);
  wcvt_kernel<<<(256 * 1152 + 255) / 256, 256, 0, stream>>>(w2, wbf2, 256 * 1152);

  // ---- layer 1: 256 -> 128
  offconv_kernel<256><<<B_ * H_, 128, 0, stream>>>(x, w_off1, b_off1, om);
  dcn_kernel<256, 128><<<dim3(B_ * H_, 1), 256, 0, stream>>>(x, om, wbf1, b1, y1);
  bnstats_kernel<128><<<128, 256, 0, stream>>>(y1, stats);
  bnapply_kernel<128><<<(B_ * 128 * HW_ + 255) / 256, 256, 0, stream>>>(y1, stats, g1, be1);

  // ---- layer 2: 128 -> 256 (pre-BN directly into d_out, normalized in place)
  offconv_kernel<128><<<B_ * H_, 128, 0, stream>>>(y1, w_off2, b_off2, om);
  dcn_kernel<128, 256><<<dim3(B_ * H_, 2), 256, 0, stream>>>(y1, om, wbf2, b2, out);
  bnstats_kernel<256><<<256, 256, 0, stream>>>(out, stats);
  bnapply_kernel<256><<<(B_ * 256 * HW_ + 255) / 256, 256, 0, stream>>>(out, stats, g2, be2);
}
